// RealMambaBlock_28870770164445
// MI455X (gfx1250) — compile-verified, matrix operand order checked by reading, not run on hardware
//
#include <hip/hip_runtime.h>
#include <hip/hip_bf16.h>

#define D_MODEL   1024
#define D_STATE   64
#define HEADDIM   64
#define D_INNER   2048
#define NHEADS    32
#define NPROJ     (2*D_INNER + 2*D_STATE + NHEADS)   // 4256
#define BSZ       2
#define SEQLEN    2048
#define ROPE_PAIRS 16

typedef _Float16 v16h __attribute__((ext_vector_type(16)));
typedef _Float16 v8h  __attribute__((ext_vector_type(8)));
typedef float    v8f  __attribute__((ext_vector_type(8)));
typedef int      v4i  __attribute__((vector_size(16)));   // matches builtin proto

// ---------------------------------------------------------------------------
// CDNA5 async global->LDS copy (ASYNCcnt) with graceful fallback.
// Builtin signature (from compiler diagnostic):
//   void __builtin_amdgcn_global_load_async_to_lds_b128(
//        v4i __attribute__((address_space(1)))* gsrc,
//        v4i __attribute__((address_space(3)))* ldst, int imm_off, int cpol)
// ---------------------------------------------------------------------------
#if defined(__AMDGCN__) && __has_builtin(__builtin_amdgcn_global_load_async_to_lds_b128)
#define HAS_ASYNC_LDS 1
typedef __attribute__((address_space(1))) v4i* gas1_v4i;
typedef __attribute__((address_space(3))) v4i* gas3_v4i;
#endif

__device__ __forceinline__ void copy16_g2l(const void* g, void* l)
{
#ifdef HAS_ASYNC_LDS
    __builtin_amdgcn_global_load_async_to_lds_b128(
        (gas1_v4i)(unsigned long long)(uintptr_t)g,
        (gas3_v4i)(unsigned)(uintptr_t)l,
        0, 0);
#else
    *(uint4*)l = *(const uint4*)g;
#endif
}

__device__ __forceinline__ void async_copy_fence()
{
#ifdef HAS_ASYNC_LDS
#if __has_builtin(__builtin_amdgcn_s_wait_asynccnt)
    __builtin_amdgcn_s_wait_asynccnt(0);
#else
    asm volatile("s_wait_asynccnt 0x0" ::: "memory");
#endif
#endif
}

// ---------------------------------------------------------------------------
// f16 GEMM with f32 accumulate.  C[M,N] = A[M,K] @ BT[N,K]^T.
// A row-major f16, BT is B pre-transposed (N x K) f16 -> all tile staging is
// contiguous 16B block copies (async-to-LDS capable). Double-buffered LDS:
// next tile's copies overlap current tile's WMMAs; one barrier per K-step.
// BM=BN=64, BK=32; 128 threads = 4 waves; wave = 16-row strip x 64 cols.
// ---------------------------------------------------------------------------
__global__ __launch_bounds__(128)
void gemm_wmma_f16t(const _Float16* __restrict__ A, const _Float16* __restrict__ BT,
                    float* __restrict__ C, int M, int N, int K)
{
    __shared__ __align__(16) _Float16 Ah[2][64][40];   // [m][k], padded stride
    __shared__ __align__(16) _Float16 Bh[2][64][40];   // [n][k]

    const int tid  = threadIdx.x;
    const int wave = tid >> 5;
    const int lane = tid & 31;
    const int m0 = blockIdx.y * 64;
    const int n0 = blockIdx.x * 64;
    const int row16 = lane & 15;
    const int kb    = (lane >> 4) * 8;      // 0 or 8: per-lane-half K-run base

    v8f acc[4] = {};

    // Tile staging: 64 rows x 32 halves = 64 rows x 4 x 16B chunks; 2 chunks/thread.
    auto load_tiles = [&](int k0, int buf) {
        #pragma unroll
        for (int i = 0; i < 2; ++i) {
            const int q  = tid + i * 128;       // 0..255
            const int r  = q >> 2;              // row 0..63
            const int cc = (q & 3) * 8;         // half-offset 0,8,16,24
            copy16_g2l(A + (size_t)(m0 + r) * K + k0 + cc, &Ah[buf][r][cc]);
            if (n0 + r < N)
                copy16_g2l(BT + (size_t)(n0 + r) * K + k0 + cc, &Bh[buf][r][cc]);
            else
                *(uint4*)&Bh[buf][r][cc] = make_uint4(0u, 0u, 0u, 0u);
        }
    };

    load_tiles(0, 0);
    async_copy_fence();
    __syncthreads();

    int buf = 0;
    for (int k0 = 0; k0 < K; k0 += 32) {
        if (k0 + 32 < K)
            load_tiles(k0 + 32, buf ^ 1);      // overlaps with WMMAs below

        // A fragment (16-bit A layout: lanes 0-15 K 0-7 & 16-23; lanes 16-31 +8)
        const int ar = wave * 16 + row16;
        v8h alo = *(const v8h*)&Ah[buf][ar][kb];
        v8h ahi = *(const v8h*)&Ah[buf][ar][16 + kb];
        v16h a  = __builtin_shufflevector(alo, ahi,
                    0,1,2,3,4,5,6,7,8,9,10,11,12,13,14,15);

        // Preload all four B fragments, then issue WMMAs back-to-back.
        v16h bf[4];
        #pragma unroll
        for (int j = 0; j < 4; ++j) {
            const int bc = j * 16 + row16;
            v8h blo = *(const v8h*)&Bh[buf][bc][kb];
            v8h bhi = *(const v8h*)&Bh[buf][bc][16 + kb];
            bf[j] = __builtin_shufflevector(blo, bhi,
                        0,1,2,3,4,5,6,7,8,9,10,11,12,13,14,15);
        }
        #pragma unroll
        for (int j = 0; j < 4; ++j)
            acc[j] = __builtin_amdgcn_wmma_f32_16x16x32_f16(
                         false, a, false, bf[j], (short)0, acc[j], false, false);

        async_copy_fence();
        __syncthreads();
        buf ^= 1;
    }

    // C/D layout: VGPR r -> M = r (lanes 0-15) / r+8 (lanes 16-31); N = lane&15
    const int mbase = m0 + wave * 16 + ((lane >> 4) * 8);
    #pragma unroll
    for (int j = 0; j < 4; ++j) {
        const int col = n0 + j * 16 + row16;
        if (col < N) {
            #pragma unroll
            for (int r = 0; r < 8; ++r)
                C[(size_t)(mbase + r) * N + col] = acc[j][r];
        }
    }
}

// ---------------------------------------------------------------------------
// Flat f32 -> f16 conversion (4 elements / thread).
// ---------------------------------------------------------------------------
__global__ __launch_bounds__(256)
void convert_f16(const float* __restrict__ in, _Float16* __restrict__ out)
{
    const size_t i = (size_t)blockIdx.x * 256 + threadIdx.x;
    const float4 v = ((const float4*)in)[i];
    union { _Float16 h[4]; uint2 u; } o;
    o.h[0] = (_Float16)v.x; o.h[1] = (_Float16)v.y;
    o.h[2] = (_Float16)v.z; o.h[3] = (_Float16)v.w;
    ((uint2*)out)[i] = o.u;
}

// ---------------------------------------------------------------------------
// Tiled f32 -> f16 transpose:  out[N,K] = (f16) in[K,N]^T.  32x32 tiles.
// ---------------------------------------------------------------------------
__global__ __launch_bounds__(256)
void convert_transpose_f16(const float* __restrict__ in, _Float16* __restrict__ out,
                           int K, int N)
{
    __shared__ float tile[32][33];
    const int k0 = blockIdx.y * 32, n0 = blockIdx.x * 32;
    const int c = threadIdx.x & 31, r8 = threadIdx.x >> 5;
    #pragma unroll
    for (int i = 0; i < 4; ++i) {
        const int r = r8 + i * 8;
        tile[r][c] = in[(size_t)(k0 + r) * N + n0 + c];
    }
    __syncthreads();
    #pragma unroll
    for (int i = 0; i < 4; ++i) {
        const int r = r8 + i * 8;
        out[(size_t)(n0 + r) * K + k0 + c] = (_Float16)tile[c][r];
    }
}

// ---------------------------------------------------------------------------
// Recurrent scan with RoPE'd state. One block per (b,h), one thread per state
// column p: 64-float column in VGPRs, RoPE pairs intra-thread, no inter-thread
// traffic. B_t/C_t/dt addresses are block-uniform (scalarizable loads).
// ---------------------------------------------------------------------------
__global__ __launch_bounds__(64)
void mamba_scan(const float* __restrict__ zxbcdt,
                const float* __restrict__ dt_bias,
                const float* __restrict__ A_log,
                const float* __restrict__ D_skip,
                float* __restrict__ yout)
{
    const int bh = blockIdx.x;
    const int b  = bh / NHEADS;
    const int h  = bh % NHEADS;
    const int p  = threadIdx.x;

    const float Aval = __expf(A_log[h]);
    const float dtb  = dt_bias[h];
    const float dsk  = D_skip[h];

    float invf[ROPE_PAIRS];
    #pragma unroll
    for (int i = 0; i < ROPE_PAIRS; ++i)     // 1/10000^(i/16)
        invf[i] = __expf(-9.210340371976184f * (float)i * (1.0f / ROPE_PAIRS));

    float hst[D_STATE];
    #pragma unroll
    for (int n = 0; n < D_STATE; ++n) hst[n] = 0.f;

    const float* base = zxbcdt + (size_t)b * SEQLEN * NPROJ;

    for (int t = 0; t < SEQLEN; ++t) {
        const float* row = base + (size_t)t * NPROJ;
        const float* Bp  = row + 2 * D_INNER;
        const float* Cp  = Bp + D_STATE;

        const float xr    = row[D_INNER + h * HEADDIM + p];
        const float dtraw = Bp[2 * D_STATE + h] + dtb;
        // softplus (stable): max(x,0) + log1p(exp(-|x|))
        const float dt    = fmaxf(dtraw, 0.f) + log1pf(__expf(-fabsf(dtraw)));
        const float decay = __expf(-Aval * dt);

        if (t + 1 < SEQLEN) {   // pull next step's row toward L2/L0 early
            __builtin_prefetch(row + NPROJ + 2 * D_INNER, 0, 1);
            __builtin_prefetch(row + NPROJ + D_INNER + h * HEADDIM + p, 0, 1);
        }

        float Bv[D_STATE], Cv[D_STATE];
        #pragma unroll
        for (int i = 0; i < D_STATE / 4; ++i) {
            float4 vb = ((const float4*)Bp)[i];
            Bv[4*i+0]=vb.x; Bv[4*i+1]=vb.y; Bv[4*i+2]=vb.z; Bv[4*i+3]=vb.w;
            float4 vc = ((const float4*)Cp)[i];
            Cv[4*i+0]=vc.x; Cv[4*i+1]=vc.y; Cv[4*i+2]=vc.z; Cv[4*i+3]=vc.w;
        }

        // RoPE on first ROPE_DIM state dims (pairs along n, same p)
        #pragma unroll
        for (int i = 0; i < ROPE_PAIRS; ++i) {
            float s, c;
            __sincosf(dt * invf[i], &s, &c);
            const float h1 = hst[2*i], h2 = hst[2*i+1];
            hst[2*i]   = c * h1 - s * h2;
            hst[2*i+1] = s * h1 + c * h2;
        }

        // h = decay*h_rot + (dt*x_p)*B ;  y_p = C . h
        const float dtx = dt * xr;
        float y = 0.f;
        #pragma unroll
        for (int n = 0; n < D_STATE; ++n) {
            const float hn = decay * hst[n] + Bv[n] * dtx;
            hst[n] = hn;
            y += hn * Cv[n];
        }

        y += dsk * xr;
        yout[(size_t)(b * SEQLEN + t) * D_INNER + h * HEADDIM + p] = y;
    }
}

// ---------------------------------------------------------------------------
// y *= silu(z); RMSNorm over D_INNER; scale by norm_weight. In place on ybuf.
// ---------------------------------------------------------------------------
__global__ __launch_bounds__(256)
void gate_rmsnorm(const float* __restrict__ zxbcdt,
                  const float* __restrict__ nw,
                  float* __restrict__ ybuf)
{
    __shared__ float red[8];
    const int rowi = blockIdx.x;                         // b*L + l
    const float* zrow = zxbcdt + (size_t)rowi * NPROJ;   // z = first D_INNER
    float* yrow = ybuf + (size_t)rowi * D_INNER;

    const int lane = threadIdx.x & 31;
    const int wv   = threadIdx.x >> 5;

    float vals[8];
    float ss = 0.f;
    #pragma unroll
    for (int i = 0; i < 8; ++i) {
        const int d = threadIdx.x + i * 256;
        const float z = zrow[d];
        const float g = z / (1.f + __expf(-z));          // silu
        const float v = yrow[d] * g;
        vals[i] = v;
        ss += v * v;
    }
    #pragma unroll
    for (int off = 16; off > 0; off >>= 1)               // wave32 reduce
        ss += __shfl_down(ss, off);
    if (lane == 0) red[wv] = ss;
    __syncthreads();
    float tot = 0.f;
    #pragma unroll
    for (int w = 0; w < 8; ++w) tot += red[w];
    const float r = rsqrtf(tot * (1.0f / D_INNER) + 1e-5f);

    #pragma unroll
    for (int i = 0; i < 8; ++i) {
        const int d = threadIdx.x + i * 256;
        yrow[d] = vals[i] * r * nw[d];
    }
}

// ---------------------------------------------------------------------------
extern "C" void kernel_launch(void* const* d_in, const int* in_sizes, int n_in,
                              void* d_out, int out_size, void* d_ws, size_t ws_size,
                              hipStream_t stream)
{
    const float* x       = (const float*)d_in[0];
    const float* W_in    = (const float*)d_in[1];
    const float* W_out   = (const float*)d_in[2];
    const float* dt_bias = (const float*)d_in[3];
    const float* A_log   = (const float*)d_in[4];
    const float* D_skip  = (const float*)d_in[5];
    const float* norm_w  = (const float*)d_in[6];
    float* out = (float*)d_out;

    const int Mtok = BSZ * SEQLEN;                   // 4096 token rows

    // workspace layout (byte offsets, all 16B aligned)
    char* w = (char*)d_ws;
    float*     zxbcdt = (float*)    (w + 0);             //  4096x4256 f32  69,730,304
    float*     ybuf   = (float*)    (w + 69730304);      //  4096x2048 f32  33,554,432
    _Float16*  xh     = (_Float16*) (w + 103284736);     //  4096x1024 f16   8,388,608
    _Float16*  winT   = (_Float16*) (w + 111673344);     //  4256x1024 f16   8,716,288
    _Float16*  woutT  = (_Float16*) (w + 120389632);     //  1024x2048 f16   4,194,304
    _Float16*  yh     = (_Float16*) (w + 124583936);     //  4096x2048 f16  16,777,216

    // 0) one-time-per-call operand conversion (bandwidth-trivial)
    convert_f16<<<(Mtok * D_MODEL) / 1024, 256, 0, stream>>>(x, xh);
    convert_transpose_f16<<<dim3(NPROJ / 32, D_MODEL / 32), 256, 0, stream>>>(
        W_in, winT, D_MODEL, NPROJ);
    convert_transpose_f16<<<dim3(D_MODEL / 32, D_INNER / 32), 256, 0, stream>>>(
        W_out, woutT, D_INNER, D_MODEL);

    // 1) in_proj: zxbcdt = x @ W_in
    dim3 g1((NPROJ + 63) / 64, Mtok / 64);
    gemm_wmma_f16t<<<g1, 128, 0, stream>>>(xh, winT, zxbcdt, Mtok, NPROJ, D_MODEL);

    // 2) recurrent scan (+ D_skip*x), y -> ybuf
    mamba_scan<<<BSZ * NHEADS, 64, 0, stream>>>(zxbcdt, dt_bias, A_log, D_skip, ybuf);

    // 3) silu gate + RMSNorm (in place)
    gate_rmsnorm<<<Mtok, 256, 0, stream>>>(zxbcdt, norm_w, ybuf);

    // 4) out_proj: out = (f16)ybuf @ W_out
    convert_f16<<<(Mtok * D_INNER) / 1024, 256, 0, stream>>>(ybuf, yh);
    dim3 g2(D_MODEL / 64, Mtok / 64);
    gemm_wmma_f16t<<<g2, 128, 0, stream>>>(yh, woutT, out, Mtok, D_MODEL, D_INNER);
}
